// CausalSelfAttention_64183991271507
// MI455X (gfx1250) — compile-verified, matrix operand order checked by reading, not run on hardware
//
#include <hip/hip_runtime.h>
#include <hip/hip_bf16.h>
#include <math.h>
#include <stdint.h>

// ---------------------------------------------------------------------------
// CausalSelfAttention for MI455X (gfx1250, wave32, WMMA).
//   B=2 T=2048 D=2048 H=16 HKV=4 DK=128
// All contractions run on v_wmma_f32_16x16x32_bf16 (bf16 in, f32 accum).
// GEMM tiles are staged memory->LDS with BOTH CDNA5 data movers:
//   - X tile: per-thread global_load_async_to_lds_b128 (ASYNCcnt)
//   - W tile: wave-level TENSOR_LOAD_TO_LDS via D# descriptor (TENSORcnt),
//     with TDM hardware padding reproducing the 40-element LDS row stride.
// Both double-buffered so the next K-slab streams while WMMAs run.
// ---------------------------------------------------------------------------

typedef __attribute__((ext_vector_type(16))) __bf16 v16bf;
typedef __attribute__((ext_vector_type(8)))  __bf16 v8bf;
typedef __attribute__((ext_vector_type(8)))  float  v8f;
typedef unsigned int v4u __attribute__((ext_vector_type(4)));
typedef int          v8i __attribute__((ext_vector_type(8)));
typedef int          v4i __attribute__((ext_vector_type(4)));

#define C_B    2
#define C_T    2048
#define C_D    2048
#define C_H    16
#define C_HKV  4
#define C_DK   128
#define C_M    (C_B * C_T)      // 4096 rows for the projection GEMMs

// Build a 16x32 (A) or 32x16 (B) bf16 fragment for one lane.
// Per ISA 7.12.2: lanes 0-15 hold K = [0..7] and [16..23]; lanes 16-31 hold
// K = [8..15] and [24..31].  Caller passes p already offset by klo = (lane>>4)*8,
// so we read p[0..7] and p[16..23].
__device__ __forceinline__ v16bf load_frag(const __bf16* p) {
    v8bf lo = *(const v8bf*)p;
    v8bf hi = *(const v8bf*)(p + 16);
    v16bf r;
#pragma unroll
    for (int i = 0; i < 8; ++i) { r[i] = lo[i]; r[i + 8] = hi[i]; }
    return r;
}

__device__ __forceinline__ v8f wmma_bf16(v16bf a, v16bf b, v8f c) {
    return __builtin_amdgcn_wmma_f32_16x16x32_bf16(
        /*neg_a=*/false, a, /*neg_b=*/false, b,
        /*c_mod=*/(short)0, c, /*reuse_a=*/false, /*reuse_b=*/false);
}

// ---- CDNA5 async memory->LDS path (ISA 15.18.3, tracked by ASYNCcnt) ------
// ldsoff: byte offset from the workgroup LDS base (low 32 bits of a generic
// pointer to __shared__). gaddr: full 64-bit global address (GV mode).
__device__ __forceinline__ void async_load_b128(uint32_t ldsoff, const void* g) {
    asm volatile("global_load_async_to_lds_b128 %0, %1, off"
                 :: "v"(ldsoff), "v"((uint64_t)(uintptr_t)g)
                 : "memory");
}
// 32B (two b128) per thread: one half-row of a 128x32 bf16 tile.
__device__ __forceinline__ void async_fill32(uint32_t ldsoff, const __bf16* g) {
    async_load_b128(ldsoff,      g);
    async_load_b128(ldsoff + 16, g + 8);
}

// ---- CDNA5 Tensor Data Mover: 128x32 bf16 tile -> LDS (ISA ch. 8) ---------
// Loads tile (tile_dim0=32 elems x tile_dim1=128 rows, data_size=2B) starting
// at gsrc with row stride `row_stride_elems`.  TDM padding inserts 4 DWORDs
// after every 16 DWORDs, so LDS rows land with stride 40 bf16 (= sW layout).
// NOTE: this toolchain exposes the 6-arg builtin
//   (uint32x4 g0, int32x8 g1, int32x4, int32x4, int32x8, i32 cpol);
// the trailing int32x8 carries descriptor words that are "unused" (=0) for a
// plain 2-D tile.
__device__ __forceinline__ void tdm_load_tile_128x32_bf16(
    uint32_t lds_off, const __bf16* gsrc, int row_stride_elems)
{
    uint64_t ga = (uint64_t)(uintptr_t)gsrc;
    v4u g0;
    g0[0] = 1u;                                   // count=1 (valid), user mode
    g0[1] = lds_off;                              // lds_addr[31:0]
    g0[2] = (uint32_t)ga;                         // global_addr[31:0]
    g0[3] = (uint32_t)(ga >> 32) | (2u << 30);    // global_addr[56:32] | type=2
    v8i g1;
    g1[0] = (1 << 16)                             // data_size: code 1 = 2 bytes
          | (1 << 20)                             // pad_enable
          | (3 << 22)                             // pad_interval: 16 DWORDs
          | (3 << 25);                            // pad_amount:    4 DWORDs
    g1[1] = (row_stride_elems & 0xFFFF) << 16;    // tensor_dim0[15:0]
    g1[2] = ((row_stride_elems >> 16) & 0xFFFF)   // tensor_dim0[31:16]
          | (128 << 16);                          // tensor_dim1[15:0] = 128
    g1[3] = (32 << 16);                           // tensor_dim1[31:16]=0 | tile_dim0=32
    g1[4] = 128;                                  // tile_dim1=128, tile_dim2=0
    g1[5] = row_stride_elems;                     // tensor_dim0_stride[31:0]
    g1[6] = 0;                                    // stride[47:32]=0
    g1[7] = 0;
    v4i g2 = {0, 0, 0, 0};                        // 2-D tile: dims 2..4 unused
    v4i g3 = {0, 0, 0, 0};
    v8i g4 = {0, 0, 0, 0, 0, 0, 0, 0};            // unused trailing words
    __builtin_amdgcn_tensor_load_to_lds(g0, g1, g2, g3, g4, /*cpol=*/0);
}

// ---------------------------------------------------------------------------
// f32 -> bf16 convert
// ---------------------------------------------------------------------------
__global__ void cvt_f32_bf16(const float* __restrict__ in,
                             __bf16* __restrict__ out, int n) {
    int i = blockIdx.x * blockDim.x + threadIdx.x;
    if (i < n) out[i] = (__bf16)in[i];
}

// ---------------------------------------------------------------------------
// GEMM: Out[m][n] = sum_k X[m][k] * W[n][k]   (X: MxK bf16, W: NxK bf16, Out f32)
// Block: 256 threads (8 waves). Tile 128x128, K-step 32, double-buffered LDS.
// X tile: async loads (all threads).  W tile: TDM (wave 0).
// Wave w computes rows [16w,16w+16) x all 128 cols (8 WMMA tiles).
// M,N multiples of 128; K multiple of 32 (true for all uses here).
// ---------------------------------------------------------------------------
__global__ __launch_bounds__(256) void gemm_xwt_bf16(
    const __bf16* __restrict__ X, const __bf16* __restrict__ W,
    float* __restrict__ Out, int M, int N, int K)
{
    __shared__ __align__(16) __bf16 sX[2][128][40];  // stride 40 (80B): 16B align kept
    __shared__ __align__(16) __bf16 sW[2][128][40];

    const int tid  = threadIdx.x;
    const int wid  = tid >> 5;
    const int lane = tid & 31;
    const int lh   = lane >> 4;
    const int ln   = lane & 15;
    const int klo  = lh * 8;
    const int m0   = blockIdx.y * 128;
    const int n0   = blockIdx.x * 128;
    const int lrow = tid >> 1;          // 128 rows, 2 threads per row
    const int lcol = (tid & 1) * 16;    // each thread moves 16 bf16 = 32B

    const __bf16* gX = X + (size_t)(m0 + lrow) * K + lcol;
    const __bf16* gW = W + (size_t)n0 * K;          // TDM reads whole W tile
    uint32_t oX[2], oW[2];
    oX[0] = (uint32_t)(uintptr_t)&sX[0][lrow][lcol];
    oX[1] = (uint32_t)(uintptr_t)&sX[1][lrow][lcol];
    oW[0] = (uint32_t)(uintptr_t)&sW[0][0][0];
    oW[1] = (uint32_t)(uintptr_t)&sW[1][0][0];

    v8f acc[8] = {};

    // prologue: stream tile 0 into buffer 0
    async_fill32(oX[0], gX);                        // 2 async b128 per thread
    if (wid == 0) tdm_load_tile_128x32_bf16(oW[0], gW, K);

    for (int k0 = 0; k0 < K; k0 += 32) {
        const int cur = (k0 >> 5) & 1;
        if (k0 + 32 < K) {               // stream next tile into the other buffer
            async_fill32(oX[cur ^ 1], gX + k0 + 32);
            if (wid == 0) tdm_load_tile_128x32_bf16(oW[cur ^ 1], gW + k0 + 32, K);
            // in-order retire: <=2 async / <=1 tensor outstanding => tile landed
            asm volatile("s_wait_asynccnt 0x2" ::: "memory");
            __builtin_amdgcn_s_wait_tensorcnt(1);
        } else {
            asm volatile("s_wait_asynccnt 0x0" ::: "memory");
            __builtin_amdgcn_s_wait_tensorcnt(0);
        }
        __syncthreads();

        v16bf a = load_frag(&sX[cur][16 * wid + ln][klo]);
#pragma unroll
        for (int j = 0; j < 8; ++j) {
            v16bf bf = load_frag(&sW[cur][16 * j + ln][klo]);
            acc[j] = wmma_bf16(a, bf, acc[j]);
        }
        __syncthreads();                 // WAR: all waves done before refill
    }

#pragma unroll
    for (int j = 0; j < 8; ++j)
#pragma unroll
        for (int r = 0; r < 8; ++r)
            Out[(size_t)(m0 + 16 * wid + 8 * lh + r) * N + (n0 + 16 * j + ln)]
                = acc[j][r];
}

// ---------------------------------------------------------------------------
// RoPE on Q (and fold in 1/sqrt(dk)); layout change [b][t][h*dk+d] -> [b][h][t][d]
// idx over B*T*H*64 (pair index p in [0,64))
// ---------------------------------------------------------------------------
__global__ void rope_q_kernel(const float* __restrict__ Qf,
                              __bf16* __restrict__ Qr) {
    int idx = blockIdx.x * blockDim.x + threadIdx.x;
    int p = idx & 63;
    int h = (idx >> 6) & (C_H - 1);
    int t = (idx >> 10) & (C_T - 1);
    int b = idx >> 21;
    float freq = __expf(-(float)(2 * p) * (9.210340371976184f / (float)C_DK)); // theta^(-2p/dk)
    float ang = (float)t * freq;
    float cs = __cosf(ang), sn = __sinf(ang);
    const float* src = Qf + ((size_t)(b * C_T + t)) * C_D + h * C_DK + 2 * p;
    float e = src[0], o = src[1];
    const float scale = 0.08838834764831845f;   // 1/sqrt(128), folded into Q
    __bf16* dst = Qr + (((size_t)b * C_H + h) * C_T + t) * C_DK + 2 * p;
    dst[0] = (__bf16)((e * cs - o * sn) * scale);
    dst[1] = (__bf16)((e * sn + o * cs) * scale);
}

// ---------------------------------------------------------------------------
// RoPE on K + transpose V:  kv f32 [b][t][1024] ->
//   Kr bf16 [b][hkv][t][dk],  Vt bf16 [b][hkv][dk][T]
// idx over B*T*HKV*64
// ---------------------------------------------------------------------------
__global__ void rope_kv_kernel(const float* __restrict__ KVf,
                               __bf16* __restrict__ Kr,
                               __bf16* __restrict__ Vt) {
    int idx = blockIdx.x * blockDim.x + threadIdx.x;
    int p  = idx & 63;
    int hk = (idx >> 6) & (C_HKV - 1);
    int t  = (idx >> 8) & (C_T - 1);
    int b  = idx >> 19;
    float freq = __expf(-(float)(2 * p) * (9.210340371976184f / (float)C_DK));
    float ang = (float)t * freq;
    float cs = __cosf(ang), sn = __sinf(ang);
    const float* src = KVf + ((size_t)(b * C_T + t)) * (2 * C_HKV * C_DK)
                       + hk * C_DK + 2 * p;
    float e = src[0], o = src[1];
    __bf16* kd = Kr + (((size_t)b * C_HKV + hk) * C_T + t) * C_DK + 2 * p;
    kd[0] = (__bf16)(e * cs - o * sn);
    kd[1] = (__bf16)(e * sn + o * cs);
    const float* vsrc = src + C_HKV * C_DK;     // v half starts at col 512
    __bf16* vd = Vt + (((size_t)b * C_HKV + hk) * C_DK + 2 * p) * C_T + t;
    vd[0]     = (__bf16)vsrc[0];
    vd[C_T]   = (__bf16)vsrc[1];
}

// ---------------------------------------------------------------------------
// Flash attention (causal, GQA). Block = 128 threads = 4 waves; each wave owns
// one 16-row query tile. K/V streamed from global (fits in 192MB L2).
// Per 32-key step: 8 QK WMMAs, online softmax, P via LDS relayout, 8 PV WMMAs.
// ---------------------------------------------------------------------------
__global__ __launch_bounds__(128) void flash_attn_kernel(
    const __bf16* __restrict__ Qr, const __bf16* __restrict__ Kr,
    const __bf16* __restrict__ Vt, __bf16* __restrict__ Attn)
{
    __shared__ __align__(16) __bf16 sP[4][16][40];

    const int lane = threadIdx.x & 31;
    const int w    = threadIdx.x >> 5;
    const int lh   = lane >> 4;
    const int ln   = lane & 15;
    const int klo  = lh * 8;
    const int qt   = blockIdx.x * 4 + w;   // 16-row query tile index
    const int h    = blockIdx.y;
    const int b    = blockIdx.z;
    const int hkv  = h >> 2;               // rep = H/HKV = 4

    const __bf16* Q  = Qr + (((size_t)b * C_H + h) * C_T + qt * 16) * C_DK;
    const __bf16* Kp = Kr + ((size_t)b * C_HKV + hkv) * C_T * C_DK;
    const __bf16* Vp = Vt + ((size_t)b * C_HKV + hkv) * C_DK * C_T;

    v16bf qa[4];
#pragma unroll
    for (int c = 0; c < 4; ++c)
        qa[c] = load_frag(Q + (size_t)ln * C_DK + 32 * c + klo);

    float mrow[8], srow[8];
    v8f acc[8] = {};
#pragma unroll
    for (int r = 0; r < 8; ++r) { mrow[r] = -1e30f; srow[r] = 0.f; }

    const int kend = qt * 16 + 15;         // causal bound
    for (int k0 = 0; k0 <= kend; k0 += 32) {
        // ---- logits: 16 q-rows x 32 k-cols (two 16x16 C fragments) ----
        v8f lg[2];
#pragma unroll
        for (int jn = 0; jn < 2; ++jn) {
            v8f cfr = {};
            const __bf16* krow = Kp + (size_t)(k0 + 16 * jn + ln) * C_DK + klo;
#pragma unroll
            for (int c = 0; c < 4; ++c) {
                v16bf kb = load_frag(krow + 32 * c);
                cfr = wmma_bf16(qa[c], kb, cfr);
            }
            lg[jn] = cfr;
        }
        // ---- online softmax (rows live across 16-lane halves) ----
        float p0[8], p1[8];
#pragma unroll
        for (int r = 0; r < 8; ++r) {
            const int row = qt * 16 + 8 * lh + r;
            float l0 = (k0 + ln)      <= row ? lg[0][r] : -1e30f;
            float l1 = (k0 + 16 + ln) <= row ? lg[1][r] : -1e30f;
            float m = fmaxf(l0, l1);
            m = fmaxf(m, __shfl_xor(m, 1));
            m = fmaxf(m, __shfl_xor(m, 2));
            m = fmaxf(m, __shfl_xor(m, 4));
            m = fmaxf(m, __shfl_xor(m, 8));
            float mnew  = fmaxf(mrow[r], m);
            float alpha = __expf(mrow[r] - mnew);
            float e0 = __expf(l0 - mnew);
            float e1 = __expf(l1 - mnew);
            float rs = e0 + e1;
            rs += __shfl_xor(rs, 1);
            rs += __shfl_xor(rs, 2);
            rs += __shfl_xor(rs, 4);
            rs += __shfl_xor(rs, 8);
            srow[r] = srow[r] * alpha + rs;
            mrow[r] = mnew;
            p0[r] = e0; p1[r] = e1;
#pragma unroll
            for (int dc = 0; dc < 8; ++dc) acc[dc][r] *= alpha;
        }
        // ---- C-layout -> A-layout for P via per-wave LDS tile ----
#pragma unroll
        for (int r = 0; r < 8; ++r) {
            sP[w][8 * lh + r][ln]      = (__bf16)p0[r];
            sP[w][8 * lh + r][16 + ln] = (__bf16)p1[r];
        }
        asm volatile("s_wait_dscnt 0x0" ::: "memory");   // intra-wave LDS RAW
        v16bf pa = load_frag(&sP[w][ln][klo]);
        // ---- PV: acc += P(16x32) x V(32x128) ----
#pragma unroll
        for (int dc = 0; dc < 8; ++dc) {
            v16bf vb = load_frag(Vp + (size_t)(16 * dc + ln) * C_T + k0 + klo);
            acc[dc] = wmma_bf16(pa, vb, acc[dc]);
        }
    }

    // ---- epilogue: normalize and scatter to [b][t][h*dk+d] (bf16) ----
    __bf16* o = Attn + ((size_t)b * C_T + qt * 16 + 8 * lh) * C_D
                + h * C_DK + ln;
#pragma unroll
    for (int dc = 0; dc < 8; ++dc)
#pragma unroll
        for (int r = 0; r < 8; ++r)
            o[(size_t)r * C_D + 16 * dc] = (__bf16)(acc[dc][r] / srow[r]);
}

// ---------------------------------------------------------------------------
// Host-side orchestration
// ---------------------------------------------------------------------------
extern "C" void kernel_launch(void* const* d_in, const int* in_sizes, int n_in,
                              void* d_out, int out_size, void* d_ws, size_t ws_size,
                              hipStream_t stream) {
    (void)in_sizes; (void)n_in; (void)out_size; (void)ws_size;
    const float* x   = (const float*)d_in[0];   // [B,T,D]
    const float* qw  = (const float*)d_in[1];   // [D,D]
    const float* kvw = (const float*)d_in[2];   // [2*HKV*DK, D]
    const float* cw  = (const float*)d_in[3];   // [D,D]
    float* out = (float*)d_out;                 // [B,T,D] f32

    char* ws = (char*)d_ws;
    size_t off = 0;
    auto carve = [&](size_t bytes) -> void* {
        void* p = ws + off;
        off += (bytes + 255) & ~(size_t)255;
        return p;
    };
    const size_t nX  = (size_t)C_B * C_T * C_D;          // 8388608
    const size_t nQW = (size_t)C_D * C_D;                // 4194304
    const size_t nKW = (size_t)2 * C_HKV * C_DK * C_D;   // 2097152

    __bf16* xb   = (__bf16*)carve(nX  * 2);
    __bf16* qwb  = (__bf16*)carve(nQW * 2);
    __bf16* kvwb = (__bf16*)carve(nKW * 2);
    __bf16* cwb  = (__bf16*)carve(nQW * 2);
    float*  qf   = (float*) carve(nX  * 4);              // Q proj, f32
    float*  kvf  = (float*) carve((size_t)C_M * 2 * C_HKV * C_DK * 4);
    __bf16* qr   = (__bf16*)carve(nX * 2);               // [b,h,t,dk]
    __bf16* kr   = (__bf16*)carve((size_t)C_B * C_HKV * C_T * C_DK * 2);
    __bf16* vt   = (__bf16*)carve((size_t)C_B * C_HKV * C_DK * C_T * 2);
    __bf16* attn = (__bf16*)carve(nX * 2);               // [b,t,D]
    // total ~124 MB of workspace

    // 1) bf16 conversions
    cvt_f32_bf16<<<(int)(nX  / 256), 256, 0, stream>>>(x,   xb,   (int)nX);
    cvt_f32_bf16<<<(int)(nQW / 256), 256, 0, stream>>>(qw,  qwb,  (int)nQW);
    cvt_f32_bf16<<<(int)(nKW / 256), 256, 0, stream>>>(kvw, kvwb, (int)nKW);
    cvt_f32_bf16<<<(int)(nQW / 256), 256, 0, stream>>>(cw,  cwb,  (int)nQW);

    // 2) projections: Q = X Qw^T, KV = X KVw^T
    gemm_xwt_bf16<<<dim3(C_D / 128, C_M / 128), 256, 0, stream>>>(
        xb, qwb, qf, C_M, C_D, C_D);
    gemm_xwt_bf16<<<dim3((2 * C_HKV * C_DK) / 128, C_M / 128), 256, 0, stream>>>(
        xb, kvwb, kvf, C_M, 2 * C_HKV * C_DK, C_D);

    // 3) RoPE + layout
    rope_q_kernel<<<(C_B * C_T * C_H * (C_DK / 2)) / 256, 256, 0, stream>>>(qf, qr);
    rope_kv_kernel<<<(C_B * C_T * C_HKV * (C_DK / 2)) / 256, 256, 0, stream>>>(kvf, kr, vt);

    // 4) causal flash attention: grid = (q-tiles/4, H, B), 4 waves per block
    flash_attn_kernel<<<dim3((C_T / 16) / 4, C_H, C_B), 128, 0, stream>>>(
        qr, kr, vt, attn);

    // 5) output projection into d_out (f32)
    gemm_xwt_bf16<<<dim3(C_D / 128, C_M / 128), 256, 0, stream>>>(
        attn, cwb, out, C_M, C_D, C_D);
}
